// PointNetSetAbstraction_36026185679269
// MI455X (gfx1250) — matrix operand order repaired and checked by reading, not compiled, not run on hardware
//
#include <hip/hip_runtime.h>
#include <hip/hip_bf16.h>

// ---------------------------------------------------------------------------
// PointNet++ Set Abstraction for MI455X (gfx1250, wave32, WMMA)
//   B=32, N=2048, D=64, NPOINT=512, RADIUS=0.2, NSAMPLE=32, MLP=[64,64,128]
// ---------------------------------------------------------------------------

#define B_   32
#define N_   2048
#define D_   64
#define S_   512
#define NS_  32
#define R2_  0.04f
#define EPS_ 1e-5f
#define M_   (B_ * S_ * NS_)      // 524288 rows of the activation matrix
#define K0PAD_ 96                 // 67 -> padded to 3 K-chunks of 32

typedef __attribute__((ext_vector_type(16))) _Float16 v16h;
typedef __attribute__((ext_vector_type(8)))  _Float16 v8h;
typedef __attribute__((ext_vector_type(8)))  float    v8f;

// ---------------------------------------------------------------------------
// 0) zero a float buffer (sums must be zero every call; harness poisons ws)
// ---------------------------------------------------------------------------
__global__ void zero_f32(float* p, int n) {
    int i = blockIdx.x * blockDim.x + threadIdx.x;
    if (i < n) p[i] = 0.0f;
}

// ---------------------------------------------------------------------------
// 1) Farthest point sampling: one block per batch. Writes new_xyz directly
//    into d_out (output 0). 256 threads, 8 points/thread, LDS argmax tree.
// ---------------------------------------------------------------------------
__global__ __launch_bounds__(256)
void fps_kernel(const float* __restrict__ xyz, float* __restrict__ out_newxyz) {
    __shared__ float sx[N_], sy[N_], sz[N_];
    __shared__ float rv[8];
    __shared__ int   ri[8];
    __shared__ int   sfar;

    const int b   = blockIdx.x;
    const int tid = threadIdx.x;

    for (int t = tid; t < N_; t += 256) {
        const float* p = xyz + ((size_t)b * N_ + t) * 3;
        sx[t] = p[0]; sy[t] = p[1]; sz[t] = p[2];
    }
    if (tid == 0) sfar = 0;

    float px[8], py[8], pz[8], dist[8];
    __syncthreads();
#pragma unroll
    for (int u = 0; u < 8; ++u) {
        int gi = u * 256 + tid;
        px[u] = sx[gi]; py[u] = sy[gi]; pz[u] = sz[gi];
        dist[u] = 1e10f;
    }

    for (int it = 0; it < S_; ++it) {
        const int far = sfar;
        const float cx = sx[far], cy = sy[far], cz = sz[far];
        if (tid == 0) {
            float* o = out_newxyz + ((size_t)b * S_ + it) * 3;
            o[0] = cx; o[1] = cy; o[2] = cz;
        }
        float bv = -1.0f; int bi = 0;
#pragma unroll
        for (int u = 0; u < 8; ++u) {
            float dx = px[u] - cx, dy = py[u] - cy, dz = pz[u] - cz;
            float d = dx * dx + dy * dy + dz * dz;
            dist[u] = fminf(dist[u], d);
            if (dist[u] > bv) { bv = dist[u]; bi = u * 256 + tid; }
        }
        // wave32 argmax (first-index tie break, matches jnp.argmax)
#pragma unroll
        for (int off = 16; off > 0; off >>= 1) {
            float ov = __shfl_xor(bv, off);
            int   oi = __shfl_xor(bi, off);
            if (ov > bv || (ov == bv && oi < bi)) { bv = ov; bi = oi; }
        }
        if ((tid & 31) == 0) { rv[tid >> 5] = bv; ri[tid >> 5] = bi; }
        __syncthreads();
        if (tid == 0) {
            float v = rv[0]; int i0 = ri[0];
#pragma unroll
            for (int w = 1; w < 8; ++w)
                if (rv[w] > v || (rv[w] == v && ri[w] < i0)) { v = rv[w]; i0 = ri[w]; }
            sfar = i0;
        }
        __syncthreads();
    }
}

// ---------------------------------------------------------------------------
// 2) Ball query + grouping: wave per centroid. Selects first NSAMPLE indices
//    (ascending) with d^2 <= r^2, pads with the first; builds f16 X0 row
//    [xyz - center (3) | points (64) | zero pad (29)] stride 96.
// ---------------------------------------------------------------------------
__global__ __launch_bounds__(256)
void group_kernel(const float* __restrict__ xyz, const float* __restrict__ points,
                  const float* __restrict__ newxyz, _Float16* __restrict__ X0) {
    __shared__ int sel[8][NS_];

    const int lane = threadIdx.x & 31;
    const int wid  = threadIdx.x >> 5;
    const int bs   = blockIdx.x * 8 + wid;           // 0 .. B*S-1
    const int b    = bs >> 9;

    const float cx = newxyz[(size_t)bs * 3 + 0];
    const float cy = newxyz[(size_t)bs * 3 + 1];
    const float cz = newxyz[(size_t)bs * 3 + 2];

    int cnt = 0;
    for (int base = 0; base < N_ && cnt < NS_; base += 32) {
        const int p = base + lane;
        const float* q = xyz + ((size_t)b * N_ + p) * 3;
        float dx = q[0] - cx, dy = q[1] - cy, dz = q[2] - cz;
        bool ok = (dx * dx + dy * dy + dz * dz) <= R2_;
        unsigned mask  = (unsigned)__ballot(ok);
        unsigned below = mask & ((1u << lane) - 1u);
        int pos = cnt + __popc(below);
        if (ok && pos < NS_) sel[wid][pos] = p;
        cnt += __popc(mask);
    }
    __syncthreads();   // block-uniform point; makes sel[] visible

    const int first = sel[wid][0];   // centroid itself guarantees cnt >= 1
    for (int j = 0; j < NS_; ++j) {
        const int pi = (j < cnt) ? sel[wid][j] : first;
        _Float16* xr = X0 + ((size_t)bs * NS_ + j) * K0PAD_;
        const float* prow = points + ((size_t)b * N_ + pi) * D_;
        xr[3 + lane]      = (_Float16)prow[lane];
        xr[3 + 32 + lane] = (_Float16)prow[lane + 32];
        if (lane < 3) {
            const float* q = xyz + ((size_t)b * N_ + pi) * 3;
            float c3 = (lane == 0) ? cx : (lane == 1) ? cy : cz;
            xr[lane] = (_Float16)(q[lane] - c3);
        }
        if (lane < 29) xr[67 + lane] = (_Float16)0.0f;
    }
}

// ---------------------------------------------------------------------------
// 3) WMMA GEMM: Y[m,n] = sum_k BNReLU(X[m,k]) * W[n,k]   (bias cancels in BN)
//    One wave per 16-row M tile, NT 16-col N tiles, KC K-chunks of 32.
//    Accumulates per-channel sum / sum^2 (LDS reduce -> global f32 atomics).
// ---------------------------------------------------------------------------
template <int NT, int KC, bool BN>
__global__ __launch_bounds__(256)
void gemm_kernel(const _Float16* __restrict__ X, const _Float16* __restrict__ W,
                 const float* __restrict__ in_scale, const float* __restrict__ in_shift,
                 _Float16* __restrict__ Y, float* __restrict__ gsum, float* __restrict__ gsq) {
    constexpr int KPAD = KC * 32;
    constexpr int COUT = NT * 16;

    __shared__ float bsum[COUT], bsq[COUT];
    if (threadIdx.x < COUT) { bsum[threadIdx.x] = 0.0f; bsq[threadIdx.x] = 0.0f; }
    __syncthreads();

    const int lane = threadIdx.x & 31;
    const int wid  = threadIdx.x >> 5;
    const size_t mtile = (size_t)blockIdx.x * 8 + wid;
    const size_t m0    = mtile * 16;
    const int mr = lane & 15;
    const int hi = (lane & 16) ? 1 : 0;              // lane-half select
    const int ak0 = hi ? 8 : 0;                      // A: K offsets {0..7}|{8..15}
    const int bk0 = hi ? 8 : 0;                      // B: K offsets {0..7}|{8..15}, +16

    const _Float16* arow = X + (m0 + mr) * KPAD;

    v8f acc[NT];
#pragma unroll
    for (int nt = 0; nt < NT; ++nt)
#pragma unroll
        for (int e = 0; e < 8; ++e) acc[nt][e] = 0.0f;

#pragma unroll
    for (int kc = 0; kc < KC; ++kc) {
        // ---- A fragment (16x32 f16): halves {ak0..ak0+7} and {ak0+16..+23}
        v16h a;
        const _Float16* ap = arow + kc * 32 + ak0;
        v8h alo = *(const v8h*)(ap);
        v8h ahi = *(const v8h*)(ap + 16);
        if (BN) {
            const int ch0 = kc * 32 + ak0;
#pragma unroll
            for (int t = 0; t < 8; ++t) {
                float f0 = fmaxf(fmaf((float)alo[t], in_scale[ch0 + t],       in_shift[ch0 + t]),       0.0f);
                float f1 = fmaxf(fmaf((float)ahi[t], in_scale[ch0 + 16 + t],  in_shift[ch0 + 16 + t]),  0.0f);
                a[t] = (_Float16)f0; a[t + 8] = (_Float16)f1;
            }
        } else {
#pragma unroll
            for (int t = 0; t < 8; ++t) { a[t] = alo[t]; a[t + 8] = ahi[t]; }
        }
#pragma unroll
        for (int nt = 0; nt < NT; ++nt) {
            // ---- B fragment (32x16 f16): col = nt*16 + (lane&15), K {bk0..+7},{bk0+16..+23}
            const _Float16* bp = W + (size_t)(nt * 16 + mr) * KPAD + kc * 32 + bk0;
            v8h b0 = *(const v8h*)(bp);
            v8h b1 = *(const v8h*)(bp + 16);
            v16h bf;
#pragma unroll
            for (int t = 0; t < 8; ++t) { bf[t] = b0[t]; bf[t + 8] = b1[t]; }
            acc[nt] = __builtin_amdgcn_wmma_f32_16x16x32_f16(
                false, a, false, bf, (short)0, acc[nt], false, false);
        }
    }

    // ---- store pre-BN f16 output + per-channel statistics
#pragma unroll
    for (int nt = 0; nt < NT; ++nt) {
        const int col = nt * 16 + mr;
        float s = 0.0f, q = 0.0f;
#pragma unroll
        for (int r = 0; r < 8; ++r) {
            float v = acc[nt][r];
            s += v; q += v * v;
            Y[(m0 + r + (hi ? 8 : 0)) * COUT + col] = (_Float16)v;
        }
        s += __shfl_xor(s, 16);
        q += __shfl_xor(q, 16);
        if (lane < 16) { atomicAdd(&bsum[col], s); atomicAdd(&bsq[col], q); }
    }
    __syncthreads();
    if (threadIdx.x < COUT) {
        atomicAdd(&gsum[threadIdx.x], bsum[threadIdx.x]);
        atomicAdd(&gsq[threadIdx.x],  bsq[threadIdx.x]);
    }
}

// ---------------------------------------------------------------------------
// 4) Fold BN stats into per-channel (scale, shift). Bias dropped (cancels).
// ---------------------------------------------------------------------------
__global__ void stats_kernel(const float* __restrict__ gsum, const float* __restrict__ gsq,
                             const float* __restrict__ g, const float* __restrict__ bt,
                             float* __restrict__ scale, float* __restrict__ shift,
                             int C, float invM) {
    int c = threadIdx.x;
    if (c < C) {
        float mean = gsum[c] * invM;
        float var  = gsq[c] * invM - mean * mean;
        float is   = rsqrtf(var + EPS_);
        float sc   = g[c] * is;
        scale[c] = sc;
        shift[c] = bt[c] - mean * sc;
    }
}

// ---------------------------------------------------------------------------
// 5) BN3 + ReLU + max over nsample -> output x (B,S,128)
// ---------------------------------------------------------------------------
__global__ __launch_bounds__(256)
void max_kernel(const _Float16* __restrict__ Y3, const float* __restrict__ scale,
                const float* __restrict__ shift, float* __restrict__ out_x) {
    const int i  = blockIdx.x * 256 + threadIdx.x;   // bs*128 + c
    const int c  = i & 127;
    const int bs = i >> 7;
    const _Float16* yb = Y3 + (size_t)bs * NS_ * 128 + c;
    const float sc = scale[c], sh = shift[c];
    float m = -1e30f;
#pragma unroll
    for (int j = 0; j < NS_; ++j)
        m = fmaxf(m, fmaf((float)yb[(size_t)j * 128], sc, sh));
    out_x[i] = fmaxf(m, 0.0f);   // relu commutes with max
}

// ---------------------------------------------------------------------------
// 6) fp32 weights -> padded f16, row-major (Cout x Kpad)
// ---------------------------------------------------------------------------
__global__ void prep_w(const float* __restrict__ w, _Float16* __restrict__ w16,
                       int Cout, int Cin, int Kpad) {
    int i = blockIdx.x * blockDim.x + threadIdx.x;
    if (i >= Cout * Kpad) return;
    int c = i / Kpad, k = i - c * Kpad;
    w16[i] = (k < Cin) ? (_Float16)w[c * Cin + k] : (_Float16)0.0f;
}

// ---------------------------------------------------------------------------
extern "C" void kernel_launch(void* const* d_in, const int* in_sizes, int n_in,
                              void* d_out, int out_size, void* d_ws, size_t ws_size,
                              hipStream_t stream) {
    (void)in_sizes; (void)n_in; (void)out_size; (void)ws_size;
    const float* xyz    = (const float*)d_in[0];
    const float* points = (const float*)d_in[1];
    const float* w0 = (const float*)d_in[2];
    const float* g0 = (const float*)d_in[4];
    const float* bt0 = (const float*)d_in[5];
    const float* w1 = (const float*)d_in[6];
    const float* g1 = (const float*)d_in[8];
    const float* bt1 = (const float*)d_in[9];
    const float* w2 = (const float*)d_in[10];
    const float* g2 = (const float*)d_in[12];
    const float* bt2 = (const float*)d_in[13];

    float* out_newxyz = (float*)d_out;                       // (B,S,3)
    float* out_x      = out_newxyz + (size_t)B_ * S_ * 3;    // (B,S,128)

    // ---- workspace carve (aligned)
    uint8_t* base = (uint8_t*)d_ws;
    size_t off = 0;
    auto alloc = [&](size_t bytes) -> void* {
        void* p = base + off;
        off += (bytes + 255) & ~(size_t)255;
        return p;
    };
    _Float16* X0  = (_Float16*)alloc((size_t)M_ * K0PAD_ * 2);
    _Float16* Y1  = (_Float16*)alloc((size_t)M_ * 64 * 2);
    _Float16* Y2  = (_Float16*)alloc((size_t)M_ * 64 * 2);
    _Float16* Y3  = (_Float16*)alloc((size_t)M_ * 128 * 2);
    _Float16* W16_0 = (_Float16*)alloc(64 * 96 * 2);
    _Float16* W16_1 = (_Float16*)alloc(64 * 64 * 2);
    _Float16* W16_2 = (_Float16*)alloc(128 * 64 * 2);
    float* fstat = (float*)alloc(1024 * sizeof(float));
    float* sum0 = fstat;        float* sq0 = fstat + 64;
    float* sum1 = fstat + 128;  float* sq1 = fstat + 192;
    float* sum2 = fstat + 256;  float* sq2 = fstat + 384;   // 128 each
    float* sc1 = fstat + 512;   float* sh1 = fstat + 576;
    float* sc2 = fstat + 640;   float* sh2 = fstat + 704;
    float* sc3 = fstat + 768;   float* sh3 = fstat + 896;   // 128 each

    const float invM = 1.0f / (float)M_;

    zero_f32<<<1, 512, 0, stream>>>(fstat, 512);
    fps_kernel<<<B_, 256, 0, stream>>>(xyz, out_newxyz);
    prep_w<<<(64 * 96 + 255) / 256, 256, 0, stream>>>(w0, W16_0, 64, 67, 96);
    prep_w<<<(64 * 64 + 255) / 256, 256, 0, stream>>>(w1, W16_1, 64, 64, 64);
    prep_w<<<(128 * 64 + 255) / 256, 256, 0, stream>>>(w2, W16_2, 128, 64, 64);
    group_kernel<<<(B_ * S_) / 8, 256, 0, stream>>>(xyz, points, out_newxyz, X0);

    // layer 0: K=96 (3 chunks), Cout=64, no input BN
    gemm_kernel<4, 3, false><<<M_ / 128, 256, 0, stream>>>(X0, W16_0, nullptr, nullptr, Y1, sum0, sq0);
    stats_kernel<<<1, 64, 0, stream>>>(sum0, sq0, g0, bt0, sc1, sh1, 64, invM);
    // layer 1: K=64 (2 chunks), Cout=64, BN+ReLU applied on A load
    gemm_kernel<4, 2, true><<<M_ / 128, 256, 0, stream>>>(Y1, W16_1, sc1, sh1, Y2, sum1, sq1);
    stats_kernel<<<1, 64, 0, stream>>>(sum1, sq1, g1, bt1, sc2, sh2, 64, invM);
    // layer 2: K=64 (2 chunks), Cout=128
    gemm_kernel<8, 2, true><<<M_ / 128, 256, 0, stream>>>(Y2, W16_2, sc2, sh2, Y3, sum2, sq2);
    stats_kernel<<<1, 128, 0, stream>>>(sum2, sq2, g2, bt2, sc3, sh3, 128, invM);

    max_kernel<<<(B_ * S_ * 128) / 256, 256, 0, stream>>>(Y3, sc3, sh3, out_x);
}
